// FlexAttentionBlock_55525337202960
// MI455X (gfx1250) — matrix operand order repaired and checked by reading, not compiled, not run on hardware
//
#include <hip/hip_runtime.h>
#include <math.h>

#define DIMX   2048
#define NHEADS 16
#define HEADD  128
#define BB     2
#define SS     2048
#define FFND   (4 * DIMX)

typedef __attribute__((ext_vector_type(16))) __bf16         v16bf;
typedef __attribute__((ext_vector_type(8)))  float          v8f;
typedef __attribute__((ext_vector_type(16))) unsigned short v16us;
typedef __attribute__((ext_vector_type(8)))  unsigned short v8us;

// ---------- helpers ----------

__device__ __forceinline__ unsigned short f2bf(float f) {
    unsigned int u = __float_as_uint(f);
    u += 0x7fffu + ((u >> 16) & 1u);      // round-to-nearest-even
    return (unsigned short)(u >> 16);
}

// A-fragment (16x32 bf16) from a row-major LDS tile: lane holds row m0+(lane&15),
// K halves: lanes 0-15 -> K 0..7 & 16..23 ; lanes 16-31 -> K 8..15 & 24..31 (ISA 7.12.2)
// Two 16B-aligned v8us loads -> 2x ds_load_b128.
__device__ __forceinline__ v16bf load_a_frag(const unsigned short* tile, int ld,
                                             int m0, int koff, int lane) {
    const unsigned short* p = tile + (size_t)(m0 + (lane & 15)) * ld + koff + ((lane >> 4) << 3);
    v8us lo = *(const v8us*)p;
    v8us hi = *(const v8us*)(p + 16);
    v16us u = __builtin_shufflevector(lo, hi, 0, 1, 2, 3, 4, 5, 6, 7,
                                              8, 9, 10, 11, 12, 13, 14, 15);
    return __builtin_bit_cast(v16bf, u);
}

// B-fragment (32x16 bf16) from an N-major ("transposed") LDS tile tileT[n][k]:
// lane holds column n0+(lane&15); lanes 0-15 -> K 0..7 & 16..23 ; lanes 16-31 -> K 8..15 & 24..31.
// Two 16B-aligned v8us loads -> 2x ds_load_b128.
__device__ __forceinline__ v16bf load_bt_frag(const unsigned short* tileT, int ld,
                                              int n0, int koff, int lane) {
    const unsigned short* p = tileT + (size_t)(n0 + (lane & 15)) * ld + koff + ((lane >> 4) << 3);
    v8us lo = *(const v8us*)p;
    v8us hi = *(const v8us*)(p + 16);
    v16us u = __builtin_shufflevector(lo, hi, 0, 1, 2, 3, 4, 5, 6, 7,
                                              8, 9, 10, 11, 12, 13, 14, 15);
    return __builtin_bit_cast(v16bf, u);
}

__device__ __forceinline__ v8f wmma_bf16(v16bf a, v16bf b, v8f c) {
    return __builtin_amdgcn_wmma_f32_16x16x32_bf16(false, a, false, b, (short)0, c, false, false);
}

// ---------- elementwise kernels ----------

__global__ __launch_bounds__(256) void cvt_bf16_kernel(const float* __restrict__ in,
                                                       unsigned short* __restrict__ out, int n) {
    int i      = blockIdx.x * 256 + threadIdx.x;
    int stride = gridDim.x * 256;
    for (; i < n; i += stride) out[i] = f2bf(in[i]);
}

__global__ __launch_bounds__(256) void rmsnorm_kernel(const float* __restrict__ x,
                                                      const float* __restrict__ wt,
                                                      unsigned short* __restrict__ out) {
    __shared__ float red[256];
    int row = blockIdx.x;
    const float* xr = x + (size_t)row * DIMX;
    float s = 0.f;
    for (int i = threadIdx.x; i < DIMX; i += 256) { float v = xr[i]; s += v * v; }
    red[threadIdx.x] = s;
    __syncthreads();
    for (int off = 128; off > 0; off >>= 1) {
        if (threadIdx.x < off) red[threadIdx.x] += red[threadIdx.x + off];
        __syncthreads();
    }
    float inv = rsqrtf(red[0] * (1.0f / DIMX) + 1e-6f);
    for (int i = threadIdx.x; i < DIMX; i += 256)
        out[(size_t)row * DIMX + i] = f2bf(xr[i] * inv * wt[i]);
}

__global__ __launch_bounds__(256) void rope_kernel(const float* __restrict__ qf,
                                                   const float* __restrict__ kf,
                                                   unsigned short* __restrict__ qb,
                                                   unsigned short* __restrict__ kb) {
    int idx = blockIdx.x * 256 + threadIdx.x;    // one thread per (row, pair)
    const int halfD = DIMX / 2;
    int row = idx / halfD;
    int p   = idx - row * halfD;                 // pair index within row: h*64 + j
    int j   = p & 63;                            // rotary freq index (head_dim/2 = 64)
    int s   = row & (SS - 1);                    // sequence position
    float ang = (float)s * __powf(10000.0f, -(float)j * (1.0f / 64.0f));
    float c = __cosf(ang), sn = __sinf(ang);
    size_t base = (size_t)row * DIMX + 2 * p;
    float qr = qf[base], qi = qf[base + 1];
    qb[base]     = f2bf(qr * c - qi * sn);
    qb[base + 1] = f2bf(qr * sn + qi * c);
    float kr = kf[base], ki = kf[base + 1];
    kb[base]     = f2bf(kr * c - ki * sn);
    kb[base + 1] = f2bf(kr * sn + ki * c);
}

// ---------- WMMA GEMM: C[M,N] = A[M,K] (bf16) x B[K,N] (bf16), f32 accum ----------
// Block tile 128x64, 8 waves in 4(M) x 2(N), each wave 32x32 = 2x2 WMMA tiles.
// EPI: 0 = store f32, 1 = store bf16, 2 = +resid store f32, 3 = SiLU store bf16

template <int EPI>
__global__ __launch_bounds__(256) void gemm_bf16_kernel(const unsigned short* __restrict__ A,
                                                        const unsigned short* __restrict__ Bw,
                                                        int M, int N, int K,
                                                        float* __restrict__ outF,
                                                        unsigned short* __restrict__ outB,
                                                        const float* __restrict__ resid) {
    __shared__ __align__(16) unsigned short la[128][40];   // A tile, row-major, padded
    __shared__ __align__(16) unsigned short lbT[64][40];   // B tile, transposed (N-major)
    const int tid  = threadIdx.x;
    const int lane = tid & 31;
    const int w    = tid >> 5;
    const int wm   = w >> 1, wn = w & 1;
    const int m0g  = blockIdx.y * 128;
    const int n0g  = blockIdx.x * 64;

    v8f zero = {0.f, 0.f, 0.f, 0.f, 0.f, 0.f, 0.f, 0.f};
    v8f acc[2][2];
    acc[0][0] = zero; acc[0][1] = zero; acc[1][0] = zero; acc[1][1] = zero;

    for (int k0 = 0; k0 < K; k0 += 32) {
        // stage A: 128 rows x 32 cols (vector copies)
#pragma unroll
        for (int l = 0; l < 2; ++l) {
            int idx = tid + l * 256;
            int r   = idx >> 2;
            int cc  = (idx & 3) << 3;
            *(uint4*)(&la[r][cc]) = *(const uint4*)(A + (size_t)(m0g + r) * K + k0 + cc);
        }
        // stage B transposed: 32(K) x 64(N) -> lbT[n][k]
        {
            int kr = tid >> 3;
            int cc = (tid & 7) << 3;
            union { uint4 v; unsigned short u[8]; } pk;
            pk.v = *(const uint4*)(Bw + (size_t)(k0 + kr) * N + n0g + cc);
#pragma unroll
            for (int i = 0; i < 8; ++i) lbT[cc + i][kr] = pk.u[i];
        }
        if (k0 + 32 < K)
            __builtin_prefetch(A + (size_t)(m0g + (tid >> 2)) * K + k0 + 32, 0, 1);
        __syncthreads();

        v16bf af0 = load_a_frag(&la[0][0], 40, wm * 32,      0, lane);
        v16bf af1 = load_a_frag(&la[0][0], 40, wm * 32 + 16, 0, lane);
        v16bf bf0 = load_bt_frag(&lbT[0][0], 40, wn * 32,      0, lane);
        v16bf bf1 = load_bt_frag(&lbT[0][0], 40, wn * 32 + 16, 0, lane);
        acc[0][0] = wmma_bf16(af0, bf0, acc[0][0]);
        acc[0][1] = wmma_bf16(af0, bf1, acc[0][1]);
        acc[1][0] = wmma_bf16(af1, bf0, acc[1][0]);
        acc[1][1] = wmma_bf16(af1, bf1, acc[1][1]);
        __syncthreads();
    }

#pragma unroll
    for (int mi = 0; mi < 2; ++mi)
#pragma unroll
        for (int ni = 0; ni < 2; ++ni)
#pragma unroll
            for (int r = 0; r < 8; ++r) {
                int m = m0g + wm * 32 + mi * 16 + r + ((lane >> 4) << 3);
                int n = n0g + wn * 32 + ni * 16 + (lane & 15);
                float val = acc[mi][ni][r];
                size_t o  = (size_t)m * N + n;
                if (EPI == 0)      outF[o] = val;
                else if (EPI == 1) outB[o] = f2bf(val);
                else if (EPI == 2) outF[o] = val + resid[o];
                else { float sg = val / (1.0f + __expf(-val)); outB[o] = f2bf(sg); }
            }
}

// ---------- causal flash attention (bf16 WMMA, f32 online softmax) ----------
// grid: (S/32, NHEADS, B); block 256 = 8 waves.
// Q tile 32 rows, KV tiles 64 rows. Scores: 8 waves = 2(M)x4(N) 16x16 tiles.
// Output: 8 waves = 2(M)x4(N) tiles of 16x32, O accum in registers.
// V is staged transposed (d-major) so P@V B-fragments are contiguous b128 loads.

__global__ __launch_bounds__(256) void attn_kernel(const unsigned short* __restrict__ q,
                                                   const unsigned short* __restrict__ k,
                                                   const unsigned short* __restrict__ v,
                                                   unsigned short* __restrict__ o) {
    __shared__ __align__(16) unsigned short lq[32][136];     // Q tile (m x d)
    __shared__ __align__(16) unsigned short lk[64][136];     // K tile (kv x d) == B^T for Q@K^T
    __shared__ __align__(16) unsigned short lvT[128][40];    // V tile transposed (d x kv)
    __shared__ float          ls[32][68];
    __shared__ __align__(16) unsigned short lp[32][72];
    __shared__ float lm[32], ll[32], lalpha[32];

    const int tid   = threadIdx.x;
    const int lane  = tid & 31;
    const int w     = tid >> 5;
    const int qbase = blockIdx.x * 32;
    const int h     = blockIdx.y;
    const int b     = blockIdx.z;
    const int dcol  = h * HEADD;
    const float scale = 0.08838834764831845f;   // 1/sqrt(128)

    // stage Q tile (32 x 128 bf16)
#pragma unroll
    for (int l = 0; l < 2; ++l) {
        int idx = tid + l * 256;
        int r   = idx >> 4;
        int cc  = (idx & 15) << 3;
        *(uint4*)(&lq[r][cc]) =
            *(const uint4*)(q + ((size_t)(b * SS + qbase + r)) * DIMX + dcol + cc);
    }
    if (tid < 32) { lm[tid] = -1e30f; ll[tid] = 0.f; }

    const int m0 = (w >> 2) * 16;
    const int n0 = (w & 3) * 32;
    v8f zero = {0.f, 0.f, 0.f, 0.f, 0.f, 0.f, 0.f, 0.f};
    v8f oacc[2]; oacc[0] = zero; oacc[1] = zero;

    for (int kb0 = 0; kb0 <= qbase + 31; kb0 += 64) {
        __syncthreads();
        // stage K row-major, V transposed (64 kv x 128 d each)
#pragma unroll
        for (int l = 0; l < 4; ++l) {
            int idx = tid + l * 256;
            int r   = idx >> 4;            // kv row 0..63
            int cc  = (idx & 15) << 3;     // d chunk 0..120
            size_t src = ((size_t)(b * SS + kb0 + r)) * DIMX + dcol + cc;
            *(uint4*)(&lk[r][cc]) = *(const uint4*)(k + src);
            union { uint4 v4; unsigned short u[8]; } pk;
            pk.v4 = *(const uint4*)(v + src);
#pragma unroll
            for (int i = 0; i < 8; ++i) lvT[cc + i][r] = pk.u[i];
        }
        __syncthreads();

        // S = Q @ K^T : wave computes a 16x16 tile at (m0s, n0s)
        const int m0s = (w >> 2) * 16;
        const int n0s = (w & 3) * 16;
        v8f sacc = zero;
#pragma unroll
        for (int d0 = 0; d0 < HEADD; d0 += 32) {
            v16bf aq = load_a_frag(&lq[0][0], 136, m0s, d0, lane);
            v16bf bk = load_bt_frag(&lk[0][0], 136, n0s, d0, lane);
            sacc = wmma_bf16(aq, bk, sacc);
        }
#pragma unroll
        for (int r = 0; r < 8; ++r) {
            int m = m0s + r + ((lane >> 4) << 3);
            int n = n0s + (lane & 15);
            ls[m][n] = sacc[r] * scale;
        }
        __syncthreads();

        // online softmax (causal) for the 32x64 score block
        if (tid < 32) {
            int row = tid;
            int qg  = qbase + row;
            float mloc = -1e30f;
            for (int c = 0; c < 64; ++c)
                if (kb0 + c <= qg) mloc = fmaxf(mloc, ls[row][c]);
            float mold = lm[row];
            float mnew = fmaxf(mold, mloc);
            float alpha = __expf(mold - mnew);
            float lsum  = ll[row] * alpha;
            for (int c = 0; c < 64; ++c) {
                float pv = (kb0 + c <= qg) ? __expf(ls[row][c] - mnew) : 0.f;
                lp[row][c] = f2bf(pv);
                lsum += pv;
            }
            lm[row] = mnew; ll[row] = lsum; lalpha[row] = alpha;
        }
        __syncthreads();

        // rescale O accumulators, then O += P @ V
#pragma unroll
        for (int ni = 0; ni < 2; ++ni)
#pragma unroll
            for (int r = 0; r < 8; ++r)
                oacc[ni][r] *= lalpha[m0 + r + ((lane >> 4) << 3)];
#pragma unroll
        for (int kk = 0; kk < 64; kk += 32) {
            v16bf ap = load_a_frag(&lp[0][0], 72, m0, kk, lane);
#pragma unroll
            for (int ni = 0; ni < 2; ++ni) {
                v16bf bv = load_bt_frag(&lvT[0][0], 40, n0 + ni * 16, kk, lane);
                oacc[ni] = wmma_bf16(ap, bv, oacc[ni]);
            }
        }
    }
    __syncthreads();

    // normalize and store (layout [b, s, h*128 + d] -> [B*S, DIM])
#pragma unroll
    for (int ni = 0; ni < 2; ++ni)
#pragma unroll
        for (int r = 0; r < 8; ++r) {
            int m = m0 + r + ((lane >> 4) << 3);
            int n = n0 + ni * 16 + (lane & 15);
            float val = oacc[ni][r] / ll[m];
            o[((size_t)(b * SS + qbase + m)) * DIMX + dcol + n] = f2bf(val);
        }
}

// ---------- launch ----------

extern "C" void kernel_launch(void* const* d_in, const int* in_sizes, int n_in,
                              void* d_out, int out_size, void* d_ws, size_t ws_size,
                              hipStream_t stream) {
    (void)in_sizes; (void)n_in; (void)out_size; (void)ws_size;

    const float* x   = (const float*)d_in[0];
    // d_in[1] is the bool triu mask; causal mask is computed inline instead.
    const float* wq  = (const float*)d_in[2];
    const float* wk  = (const float*)d_in[3];
    const float* wv  = (const float*)d_in[4];
    const float* wo  = (const float*)d_in[5];
    const float* w1  = (const float*)d_in[6];
    const float* w2  = (const float*)d_in[7];
    const float* anw = (const float*)d_in[8];
    const float* fnw = (const float*)d_in[9];
    float* out = (float*)d_out;

    const int D = DIMX, M = BB * SS, F = FFND;
    char* ws = (char*)d_ws;
    size_t off = 0;
    auto alloc = [&](size_t bytes) { size_t o = off; off += (bytes + 255) & ~(size_t)255; return o; };

    unsigned short* wq_b = (unsigned short*)(ws + alloc((size_t)D * D * 2));
    unsigned short* wk_b = (unsigned short*)(ws + alloc((size_t)D * D * 2));
    unsigned short* wv_b = (unsigned short*)(ws + alloc((size_t)D * D * 2));
    unsigned short* wo_b = (unsigned short*)(ws + alloc((size_t)D * D * 2));
    unsigned short* w1_b = (unsigned short*)(ws + alloc((size_t)D * F * 2));
    unsigned short* w2_b = (unsigned short*)(ws + alloc((size_t)F * D * 2));
    unsigned short* h_b  = (unsigned short*)(ws + alloc((size_t)M * D * 2));  // reused for h2
    float*          qf   = (float*)(ws + alloc((size_t)M * D * 4));
    float*          kf   = (float*)(ws + alloc((size_t)M * D * 4));
    unsigned short* qb_  = (unsigned short*)(ws + alloc((size_t)M * D * 2));
    unsigned short* kb_  = (unsigned short*)(ws + alloc((size_t)M * D * 2));
    unsigned short* vb_  = (unsigned short*)(ws + alloc((size_t)M * D * 2));
    unsigned short* at_b = (unsigned short*)(ws + alloc((size_t)M * D * 2));
    float*          x1   = (float*)(ws + alloc((size_t)M * D * 4));
    unsigned short* ffn1_b = (unsigned short*)qf;  // qf+kf region reused (64 MB) after RoPE

    // 1) weights -> bf16
    cvt_bf16_kernel<<<2048, 256, 0, stream>>>(wq, wq_b, D * D);
    cvt_bf16_kernel<<<2048, 256, 0, stream>>>(wk, wk_b, D * D);
    cvt_bf16_kernel<<<2048, 256, 0, stream>>>(wv, wv_b, D * D);
    cvt_bf16_kernel<<<2048, 256, 0, stream>>>(wo, wo_b, D * D);
    cvt_bf16_kernel<<<4096, 256, 0, stream>>>(w1, w1_b, D * F);
    cvt_bf16_kernel<<<4096, 256, 0, stream>>>(w2, w2_b, F * D);

    // 2) h = rmsnorm(x) (bf16)
    rmsnorm_kernel<<<M, 256, 0, stream>>>(x, anw, h_b);

    // 3) q,k (f32 for RoPE), v (bf16)
    dim3 gD(D / 64, M / 128);
    gemm_bf16_kernel<0><<<gD, 256, 0, stream>>>(h_b, wq_b, M, D, D, qf, nullptr, nullptr);
    gemm_bf16_kernel<0><<<gD, 256, 0, stream>>>(h_b, wk_b, M, D, D, kf, nullptr, nullptr);
    gemm_bf16_kernel<1><<<gD, 256, 0, stream>>>(h_b, wv_b, M, D, D, nullptr, vb_, nullptr);

    // 4) RoPE -> bf16 q,k
    rope_kernel<<<(M * (D / 2)) / 256, 256, 0, stream>>>(qf, kf, qb_, kb_);

    // 5) attention
    dim3 gA(SS / 32, NHEADS, BB);
    attn_kernel<<<gA, 256, 0, stream>>>(qb_, kb_, vb_, at_b);

    // 6) x1 = x + attn @ wo
    gemm_bf16_kernel<2><<<gD, 256, 0, stream>>>(at_b, wo_b, M, D, D, x1, nullptr, x);

    // 7) h2 = rmsnorm(x1)
    rmsnorm_kernel<<<M, 256, 0, stream>>>(x1, fnw, h_b);

    // 8) ffn1 = silu(h2 @ w1)
    dim3 gF1(F / 64, M / 128);
    gemm_bf16_kernel<3><<<gF1, 256, 0, stream>>>(h_b, w1_b, M, F, D, nullptr, ffn1_b, nullptr);

    // 9) out = x1 + ffn1 @ w2
    gemm_bf16_kernel<2><<<gD, 256, 0, stream>>>(ffn1_b, w2_b, M, D, F, out, nullptr, x1);
}
